// GPT_50706383897048
// MI455X (gfx1250) — compile-verified, matrix operand order checked by reading, not run on hardware
//
#include <hip/hip_runtime.h>
#include <hip/hip_bf16.h>
#include <math.h>

typedef __attribute__((ext_vector_type(16))) _Float16 v16h;
typedef __attribute__((ext_vector_type(8)))  float    v8f;
typedef __attribute__((ext_vector_type(4)))  unsigned int v4u;
typedef __attribute__((ext_vector_type(8)))  int      v8i;
typedef __attribute__((ext_vector_type(4)))  int      v4i;

#define NB 8
#define NT 1024
#define NE 768
#define NH 12
#define NL 12
#define NV 50304
#define NE3 2304
#define NE4 3072
#define NTOK (NB * NT)

static __device__ __forceinline__ v8f wmma_f16(v16h a, v16h b, v8f c) {
    return __builtin_amdgcn_wmma_f32_16x16x32_f16(false, a, false, b, (short)0, c,
                                                  false, false);
}

// ---------------------------------------------------------------------------
// TDM: issue a 2-D f16 tile load (tile_k x tile_rows, row stride = stride0
// elements) from global memory into LDS at byte offset lds_off.
// D# layout per CDNA5 ISA (groups 2/3 zero: 2-D tile).
// This toolchain exposes the 6-arg builtin:
//   (uint32x4 g0, int32x8 g1, int32x4 g2, int32x4 g3, int32x8, i32 cpol)
// ---------------------------------------------------------------------------
static __device__ __forceinline__ void tdm_load_2d(unsigned lds_off,
                                                   const _Float16* gptr,
                                                   unsigned tile_k,
                                                   unsigned tile_rows,
                                                   unsigned tensor_d0,
                                                   unsigned tensor_d1,
                                                   unsigned stride0) {
    unsigned long long ga = (unsigned long long)(const void*)gptr;
    v4u g0;
    g0.x = 1u;                                   // count=1, user-mode descriptor
    g0.y = lds_off;                              // LDS byte address
    g0.z = (unsigned)(ga & 0xFFFFFFFFu);         // global_addr[31:0]
    g0.w = (unsigned)((ga >> 32) & 0x01FFFFFFu)  // global_addr[56:32]
           | (2u << 30);                         // type = 2 ("image")
    v8i g1;
    g1[0] = (int)(1u << 16);                               // data_size=1 (2B)
    g1[1] = (int)((tensor_d0 & 0xFFFFu) << 16);            // tensor_dim0[15:0]
    g1[2] = (int)((tensor_d0 >> 16) | ((tensor_d1 & 0xFFFFu) << 16));
    g1[3] = (int)((tensor_d1 >> 16) | (tile_k << 16));     // tile_dim0
    g1[4] = (int)(tile_rows & 0xFFFFu);                    // tile_dim1, tile_dim2=0
    g1[5] = (int)stride0;                                  // tensor_dim0_stride[31:0]
    g1[6] = 0;                                             // stride0[47:32]|d1_stride lo
    g1[7] = 0;
    v4i z4 = {0, 0, 0, 0};
    v8i z8 = {0, 0, 0, 0, 0, 0, 0, 0};
    __builtin_amdgcn_tensor_load_to_lds(g0, g1, z4, z4, z8, 0);
}

// ---------------------------------------------------------------------------
// fp32 -> f16 weight conversion (grid-stride)
// ---------------------------------------------------------------------------
__global__ void f32_to_f16_kernel(const float* __restrict__ src,
                                  _Float16* __restrict__ dst, size_t n) {
    size_t i = (size_t)blockIdx.x * blockDim.x + threadIdx.x;
    size_t stride = (size_t)gridDim.x * blockDim.x;
    for (; i < n; i += stride) dst[i] = (_Float16)src[i];
}

// ---------------------------------------------------------------------------
// x[b,t,:] = wte[idx[b,t],:] + wpe[t,:]   (fp32 residual stream)
// ---------------------------------------------------------------------------
__global__ void embed_kernel(const int* __restrict__ idx,
                             const float* __restrict__ wte,
                             const float* __restrict__ wpe,
                             float* __restrict__ x) {
    int tok = blockIdx.x;
    int t = tok % NT;
    int id = idx[tok];
    const float* we = wte + (size_t)id * NE;
    const float* wp = wpe + (size_t)t * NE;
    float* xr = x + (size_t)tok * NE;
#pragma unroll
    for (int j = 0; j < 3; ++j) {
        int c = threadIdx.x + j * 256;
        xr[c] = we[c] + wp[c];
    }
}

// ---------------------------------------------------------------------------
// LayerNorm over E=768, fp32 in -> f16 out. One block (256 thr) per token.
// ---------------------------------------------------------------------------
__global__ void ln_kernel(const float* __restrict__ x,
                          const float* __restrict__ w,
                          const float* __restrict__ b,
                          _Float16* __restrict__ out) {
    __shared__ float s1[256], s2[256];
    int tok = blockIdx.x;
    const float* xr = x + (size_t)tok * NE;
    float v0 = xr[threadIdx.x];
    float v1 = xr[threadIdx.x + 256];
    float v2 = xr[threadIdx.x + 512];
    s1[threadIdx.x] = v0 + v1 + v2;
    s2[threadIdx.x] = v0 * v0 + v1 * v1 + v2 * v2;
    __syncthreads();
    for (int off = 128; off > 0; off >>= 1) {
        if (threadIdx.x < off) {
            s1[threadIdx.x] += s1[threadIdx.x + off];
            s2[threadIdx.x] += s2[threadIdx.x + off];
        }
        __syncthreads();
    }
    float mu = s1[0] * (1.0f / NE);
    float var = s2[0] * (1.0f / NE) - mu * mu;
    float rs = rsqrtf(var + 1e-5f);
    _Float16* orow = out + (size_t)tok * NE;
#pragma unroll
    for (int j = 0; j < 3; ++j) {
        int c = threadIdx.x + j * 256;
        float v = (j == 0) ? v0 : (j == 1) ? v1 : v2;
        orow[c] = (_Float16)((v - mu) * rs * w[c] + b[c]);
    }
}

// ---------------------------------------------------------------------------
// Final LN only on the 8 last-position rows, into a 64-row zero-padded f16
// buffer (rows 8..63 zeroed) so the lm_head GEMM needs no load guards.
// ---------------------------------------------------------------------------
__global__ void lastln_kernel(const float* __restrict__ x,
                              const float* __restrict__ w,
                              const float* __restrict__ b,
                              _Float16* __restrict__ out) {
    __shared__ float s1[256], s2[256];
    int row = blockIdx.x;  // 0..63
    _Float16* orow = out + (size_t)row * NE;
    if (row >= NB) {
#pragma unroll
        for (int j = 0; j < 3; ++j) orow[threadIdx.x + j * 256] = (_Float16)0.0f;
        return;
    }
    int tok = row * NT + (NT - 1);
    const float* xr = x + (size_t)tok * NE;
    float v0 = xr[threadIdx.x];
    float v1 = xr[threadIdx.x + 256];
    float v2 = xr[threadIdx.x + 512];
    s1[threadIdx.x] = v0 + v1 + v2;
    s2[threadIdx.x] = v0 * v0 + v1 * v1 + v2 * v2;
    __syncthreads();
    for (int off = 128; off > 0; off >>= 1) {
        if (threadIdx.x < off) {
            s1[threadIdx.x] += s1[threadIdx.x + off];
            s2[threadIdx.x] += s2[threadIdx.x + off];
        }
        __syncthreads();
    }
    float mu = s1[0] * (1.0f / NE);
    float var = s2[0] * (1.0f / NE) - mu * mu;
    float rs = rsqrtf(var + 1e-5f);
#pragma unroll
    for (int j = 0; j < 3; ++j) {
        int c = threadIdx.x + j * 256;
        float v = (j == 0) ? v0 : (j == 1) ? v1 : v2;
        orow[c] = (_Float16)((v - mu) * rs * w[c] + b[c]);
    }
}

// ---------------------------------------------------------------------------
// Gather V^T:  vT[b,h,d,t] = qkv16[(b*T+t)*3E + 2E + h*64 + d]
// ---------------------------------------------------------------------------
__global__ void transpose_v_kernel(const _Float16* __restrict__ qkv,
                                   _Float16* __restrict__ vT) {
    size_t total = (size_t)NB * NH * 64 * NT;
    size_t stride = (size_t)gridDim.x * blockDim.x;
    for (size_t o = (size_t)blockIdx.x * blockDim.x + threadIdx.x; o < total;
         o += stride) {
        int t = (int)(o & (NT - 1));
        int d = (int)((o >> 10) & 63);
        int rest = (int)(o >> 16);
        int h = rest % NH;
        int bq = rest / NH;
        vT[o] = qkv[(size_t)(bq * NT + t) * NE3 + 2 * NE + h * 64 + d];
    }
}

// ---------------------------------------------------------------------------
// WMMA GEMM with TDM-staged tiles:  C[M,N] = A[M,K] (f16) * W[N,K]^T + bias
// Block = 256 thr = 8 waves; wave -> 16x64 strip; block tile 64x128, K step 32.
// Wave 0 DMAs the 64x32 A tile and 128x32 B tile into LDS via
// tensor_load_to_lds (double-buffered, s_wait_tensorcnt + barrier handoff);
// all waves compute from LDS. Dynamic LDS only, so descriptor lds_addr
// offsets are workgroup-relative: A0@0, A1@4K, B0@8K, B1@16K (24KB total).
// MODE 0: f16 store          MODE 1: f16 store with exact GELU
// MODE 2: fp32 residual add  MODE 3: fp32 store, row-guarded (lm_head)
// ---------------------------------------------------------------------------
template <int MODE>
__global__ __launch_bounds__(256) void gemm16_kernel(
    const _Float16* __restrict__ A, const _Float16* __restrict__ W,
    const float* __restrict__ bias, void* __restrict__ outp,
    const float* __restrict__ res, int M, int N, int K, int Mstore) {
    extern __shared__ _Float16 smem[];  // halves: A0@0 A1@2048 B0@4096 B1@8192
    int wave = threadIdx.x >> 5;
    int lane = threadIdx.x & 31;
    int lrow = lane & 15;
    int lhalf = lane >> 4;
    int m0 = blockIdx.x * 64;
    int n0 = blockIdx.y * 128;

    v8f acc[4];
#pragma unroll
    for (int s = 0; s < 4; ++s) acc[s] = (v8f){0, 0, 0, 0, 0, 0, 0, 0};

    const int nsteps = K >> 5;
    if (wave == 0) {
        tdm_load_2d(0u, A + (size_t)m0 * K, 32u, 64u, (unsigned)K, (unsigned)M,
                    (unsigned)K);
        tdm_load_2d(8192u, W + (size_t)n0 * K, 32u, 128u, (unsigned)K,
                    (unsigned)N, (unsigned)K);
    }

    for (int i = 0; i < nsteps; ++i) {
        int sel = i & 1;
        if (wave == 0) __builtin_amdgcn_s_wait_tensorcnt((short)0);
        __syncthreads();  // staged tiles visible to all waves
        if (wave == 0 && (i + 1) < nsteps) {
            int k = (i + 1) << 5;
            tdm_load_2d(sel ? 0u : 4096u, A + (size_t)m0 * K + k, 32u, 64u,
                        (unsigned)K, (unsigned)M, (unsigned)K);
            tdm_load_2d(sel ? 8192u : 16384u, W + (size_t)n0 * K + k, 32u, 128u,
                        (unsigned)K, (unsigned)N, (unsigned)K);
        }
        const _Float16* As =
            smem + (sel ? 2048 : 0) + ((wave & 3) * 16 + lrow) * 32 + lhalf * 16;
        const _Float16* Bs = smem + (sel ? 8192 : 4096) +
                             ((wave >> 2) * 64 + lrow) * 32 + lhalf * 16;
        v16h a = *(const v16h*)As;
#pragma unroll
        for (int s = 0; s < 4; ++s) {
            v16h bm = *(const v16h*)(Bs + s * 512);  // +16 rows per subtile
            acc[s] = wmma_f16(a, bm, acc[s]);
        }
        __syncthreads();  // all reads done before next overwrite of this buffer
    }

#pragma unroll
    for (int s = 0; s < 4; ++s) {
        int col = n0 + (wave >> 2) * 64 + s * 16 + lrow;
        float bv = bias[col];
#pragma unroll
        for (int r = 0; r < 8; ++r) {
            int row = m0 + (wave & 3) * 16 + r + lhalf * 8;
            if (row < Mstore) {
                float val = acc[s][r] + bv;
                size_t oi = (size_t)row * N + col;
                if (MODE == 0) {
                    ((_Float16*)outp)[oi] = (_Float16)val;
                } else if (MODE == 1) {
                    float g = 0.5f * val * (1.0f + erff(val * 0.70710678118f));
                    ((_Float16*)outp)[oi] = (_Float16)g;
                } else if (MODE == 2) {
                    ((float*)outp)[oi] = res[oi] + val;
                } else {
                    ((float*)outp)[oi] = val;
                }
            }
        }
    }
}

// ---------------------------------------------------------------------------
// Causal attention for one (b, h, 16-query tile).
// scores = q.k^T (no 1/sqrt(d) scaling, faithful), causal mask, fp32 softmax
// (register-resident scores, cross-wave stats via LDS), y = P.V via WMMA
// against pre-transposed V^T, scaled by 1/rowsum at the end.
// qkv layout per token: [k(0:E) | q(E:2E) | v(2E:3E)]   (k first!)
// ---------------------------------------------------------------------------
__global__ __launch_bounds__(256) void attn_kernel(
    const _Float16* __restrict__ qkv, const _Float16* __restrict__ vT,
    _Float16* __restrict__ y) {
    int qt = blockIdx.x & 63;
    int rest = blockIdx.x >> 6;
    int h = rest % NH;
    int bq = rest / NH;

    int wave = threadIdx.x >> 5;
    int lane = threadIdx.x & 31;
    int lrow = lane & 15;
    int lhalf = lane >> 4;

    __shared__ __align__(32) _Float16 probs[16][NT];
    __shared__ float redmax[8][16];
    __shared__ float redsum[8][16];
    __shared__ float rowmax[16];
    __shared__ float rowsum[16];

    const _Float16* qbase = qkv + (size_t)(bq * NT + qt * 16 + lrow) * NE3 +
                            NE + h * 64 + lhalf * 16;
    v16h qa0 = *(const v16h*)(qbase);
    v16h qa1 = *(const v16h*)(qbase + 32);

    v8f acc[8];
#pragma unroll
    for (int i = 0; i < 8; ++i) acc[i] = (v8f){0, 0, 0, 0, 0, 0, 0, 0};
#pragma unroll
    for (int i = 0; i < 8; ++i) {
        int kt = wave + 8 * i;
        if (kt <= qt) {
            const _Float16* kb = qkv + (size_t)(bq * NT + kt * 16 + lrow) * NE3 +
                                 h * 64 + lhalf * 16;
            v16h kb0 = *(const v16h*)(kb);
            v16h kb1 = *(const v16h*)(kb + 32);
            acc[i] = wmma_f16(qa0, kb0, acc[i]);
            acc[i] = wmma_f16(qa1, kb1, acc[i]);
        }
    }
#pragma unroll
    for (int i = 0; i < 8; ++i) {
        int kt = wave + 8 * i;
        if (kt == qt) {
#pragma unroll
            for (int r = 0; r < 8; ++r) {
                int R = r + lhalf * 8;
                if (lrow > R) acc[i][r] = -3.0e38f;
            }
        }
    }

#pragma unroll
    for (int r = 0; r < 8; ++r) {
        float m = -3.0e38f;
#pragma unroll
        for (int i = 0; i < 8; ++i) {
            int kt = wave + 8 * i;
            if (kt <= qt) m = fmaxf(m, acc[i][r]);
        }
        for (int off = 8; off >= 1; off >>= 1) m = fmaxf(m, __shfl_xor(m, off, 16));
        if (lrow == 0) redmax[wave][r + lhalf * 8] = m;
    }
    __syncthreads();
    if (threadIdx.x < 16) {
        float m = -3.0e38f;
#pragma unroll
        for (int w = 0; w < 8; ++w) m = fmaxf(m, redmax[w][threadIdx.x]);
        rowmax[threadIdx.x] = m;
    }
    __syncthreads();

#pragma unroll
    for (int r = 0; r < 8; ++r) {
        int R = r + lhalf * 8;
        float rm = rowmax[R];
        float s = 0.0f;
#pragma unroll
        for (int i = 0; i < 8; ++i) {
            int kt = wave + 8 * i;
            if (kt <= qt) {
                float e = __expf(acc[i][r] - rm);
                probs[R][kt * 16 + lrow] = (_Float16)e;
                s += e;
            }
        }
        for (int off = 8; off >= 1; off >>= 1) s += __shfl_xor(s, off, 16);
        if (lrow == 0) redsum[wave][R] = s;
    }
    int ncols = (qt + 1) * 16;
    int padcols = (ncols + 31) & ~31;
    if (padcols > ncols) {
        int rr = threadIdx.x >> 4;
        int cc = ncols + (threadIdx.x & 15);
        if (rr < 16) probs[rr][cc] = (_Float16)0.0f;
    }
    __syncthreads();
    if (threadIdx.x < 16) {
        float s = 0.0f;
#pragma unroll
        for (int w = 0; w < 8; ++w) s += redsum[w][threadIdx.x];
        rowsum[threadIdx.x] = s;
    }
    __syncthreads();

    if (wave < 4) {
        v8f accy = (v8f){0, 0, 0, 0, 0, 0, 0, 0};
        const _Float16* vrow =
            vT + (size_t)((bq * NH + h) * 64 + wave * 16 + lrow) * NT + lhalf * 16;
        for (int k0 = 0; k0 < padcols; k0 += 32) {
            v16h a = *(const v16h*)(&probs[lrow][k0 + lhalf * 16]);
            v16h bm = *(const v16h*)(vrow + k0);
            accy = wmma_f16(a, bm, accy);
        }
#pragma unroll
        for (int r = 0; r < 8; ++r) {
            int R = r + lhalf * 8;
            float val = accy[r] / rowsum[R];
            y[(size_t)(bq * NT + qt * 16 + R) * NE + h * 64 + wave * 16 + lrow] =
                (_Float16)val;
        }
    }
}

// ---------------------------------------------------------------------------
// Host orchestration
// ---------------------------------------------------------------------------
static inline size_t alignup(size_t v) { return (v + 255) & ~(size_t)255; }

#define GEMM_LDS 24576

extern "C" void kernel_launch(void* const* d_in, const int* in_sizes, int n_in,
                              void* d_out, int out_size, void* d_ws, size_t ws_size,
                              hipStream_t stream) {
    const int* idx = (const int*)d_in[0];
    const float* wte = (const float*)d_in[1];
    const float* wpe = (const float*)d_in[2];
    const float* qkv_w = (const float*)d_in[3];
    const float* qkv_b = (const float*)d_in[4];
    const float* proj_w = (const float*)d_in[5];
    const float* proj_b = (const float*)d_in[6];
    const float* ln_w = (const float*)d_in[7];
    const float* ln_b = (const float*)d_in[8];
    const float* c1_w = (const float*)d_in[9];
    const float* c1_b = (const float*)d_in[10];
    const float* c2_w = (const float*)d_in[11];
    const float* c2_b = (const float*)d_in[12];
    const float* lnf_w = (const float*)d_in[13];
    const float* lnf_b = (const float*)d_in[14];
    const float* lm_b = (const float*)d_in[15];
    float* out = (float*)d_out;

    char* ws = (char*)d_ws;
    size_t off = 0;
    float* x = (float*)(ws + off);            off += alignup((size_t)NTOK * NE * 4);
    _Float16* h16 = (_Float16*)(ws + off);    off += alignup((size_t)NTOK * NE * 2);
    _Float16* qkv16 = (_Float16*)(ws + off);  off += alignup((size_t)NTOK * NE3 * 2);
    _Float16* vT16 = (_Float16*)(ws + off);   off += alignup((size_t)NB * NH * 64 * NT * 2);
    _Float16* y16 = (_Float16*)(ws + off);    off += alignup((size_t)NTOK * NE * 2);
    _Float16* mlp16 = (_Float16*)(ws + off);  off += alignup((size_t)NTOK * NE4 * 2);
    _Float16* pad16 = (_Float16*)(ws + off);  off += alignup((size_t)64 * NE * 2);
    _Float16* wte16 = (_Float16*)(ws + off);  off += alignup((size_t)NV * NE * 2);
    _Float16* wqkv16 = (_Float16*)(ws + off); off += alignup((size_t)NL * NE3 * NE * 2);
    _Float16* wproj16 = (_Float16*)(ws + off);off += alignup((size_t)NL * NE * NE * 2);
    _Float16* wc116 = (_Float16*)(ws + off);  off += alignup((size_t)NL * NE4 * NE * 2);
    _Float16* wc216 = (_Float16*)(ws + off);  off += alignup((size_t)NL * NE * NE4 * 2);
    (void)ws_size; (void)n_in; (void)in_sizes; (void)out_size;

    dim3 cvtgrid(2048), cvtblk(256);
    f32_to_f16_kernel<<<cvtgrid, cvtblk, 0, stream>>>(wte, wte16, (size_t)NV * NE);
    f32_to_f16_kernel<<<cvtgrid, cvtblk, 0, stream>>>(qkv_w, wqkv16, (size_t)NL * NE3 * NE);
    f32_to_f16_kernel<<<cvtgrid, cvtblk, 0, stream>>>(proj_w, wproj16, (size_t)NL * NE * NE);
    f32_to_f16_kernel<<<cvtgrid, cvtblk, 0, stream>>>(c1_w, wc116, (size_t)NL * NE4 * NE);
    f32_to_f16_kernel<<<cvtgrid, cvtblk, 0, stream>>>(c2_w, wc216, (size_t)NL * NE * NE4);

    embed_kernel<<<dim3(NTOK), dim3(256), 0, stream>>>(idx, wte, wpe, x);

    for (int l = 0; l < NL; ++l) {
        const float* lw = ln_w + (size_t)l * NE;
        const float* lb = ln_b + (size_t)l * NE;

        ln_kernel<<<dim3(NTOK), dim3(256), 0, stream>>>(x, lw, lb, h16);
        gemm16_kernel<0><<<dim3(NTOK / 64, NE3 / 128), dim3(256), GEMM_LDS, stream>>>(
            h16, wqkv16 + (size_t)l * NE3 * NE, qkv_b + (size_t)l * NE3,
            (void*)qkv16, nullptr, NTOK, NE3, NE, NTOK);
        transpose_v_kernel<<<dim3(2048), dim3(256), 0, stream>>>(qkv16, vT16);
        attn_kernel<<<dim3(NB * NH * (NT / 16)), dim3(256), 0, stream>>>(qkv16, vT16, y16);
        gemm16_kernel<2><<<dim3(NTOK / 64, NE / 128), dim3(256), GEMM_LDS, stream>>>(
            y16, wproj16 + (size_t)l * NE * NE, proj_b + (size_t)l * NE,
            (void*)x, x, NTOK, NE, NE, NTOK);
        ln_kernel<<<dim3(NTOK), dim3(256), 0, stream>>>(x, lw, lb, h16);
        gemm16_kernel<1><<<dim3(NTOK / 64, NE4 / 128), dim3(256), GEMM_LDS, stream>>>(
            h16, wc116 + (size_t)l * NE4 * NE, c1_b + (size_t)l * NE4,
            (void*)mlp16, nullptr, NTOK, NE4, NE, NTOK);
        gemm16_kernel<2><<<dim3(NTOK / 64, NE / 128), dim3(256), GEMM_LDS, stream>>>(
            mlp16, wc216 + (size_t)l * NE * NE4, c2_b + (size_t)l * NE,
            (void*)x, x, NTOK, NE, NE4, NTOK);
    }

    lastln_kernel<<<dim3(64), dim3(256), 0, stream>>>(x, lnf_w, lnf_b, pad16);
    gemm16_kernel<3><<<dim3(1, NV / 128), dim3(256), GEMM_LDS, stream>>>(
        pad16, wte16, lm_b, (void*)out, nullptr, 64, NV, NE, NB);
}
